// CustomOrientedRCNN_64819646431445
// MI455X (gfx1250) — compile-verified, matrix operand order checked by reading, not run on hardware
//
#include <hip/hip_runtime.h>
#include <cstdint>
#include <cstddef>

// ---------------------------------------------------------------------------
// CustomOrientedRCNN on MI455X (gfx1250, wave32, WMMA)
//
// Key structural facts exploited:
//  * obj logits are dead code (never returned)
//  * proposals[:, :100] => only (y=0, x=0..3) RPN/delta columns are live
//  * fc1 (800x12544 @ 12544x1024) dominates FLOPs -> bf16 WMMA, f32 accum
//  * GEMM register-blocked 2x4 (32x64 out per wave): 24 B/lane per WMMA
//    instead of 64 B/lane -> lifts the loop off the L2 bandwidth ceiling
// ---------------------------------------------------------------------------

typedef __attribute__((ext_vector_type(16))) __bf16 v16bf;
typedef __attribute__((ext_vector_type(8)))  __bf16 v8bf;
typedef __attribute__((ext_vector_type(8)))  float  v8f;

__device__ __forceinline__ unsigned short f32_to_bf16_bits(float f) {
  unsigned int u = __float_as_uint(f);
  u = u + 0x7FFFu + ((u >> 16) & 1u);      // round-to-nearest-even
  return (unsigned short)(u >> 16);
}

// --------------------------- fp32 -> bf16 convert ---------------------------
__global__ void __launch_bounds__(256)
cvt_bf16_kernel(const float* __restrict__ in, unsigned short* __restrict__ out, int n) {
  int i = blockIdx.x * blockDim.x + threadIdx.x;
  int stride = gridDim.x * blockDim.x;
  for (; i < n; i += stride) out[i] = f32_to_bf16_bits(in[i]);
}

// ------------------- backbone stride-16 conv + 2x2 avgpool ------------------
// One block per (b, y, x) pooled pixel; thread = output channel c (256).
// The four conv taps share weights, so pre-sum the four 3x3x3 patches.
__global__ void __launch_bounds__(256)
backbone_pool_kernel(const float* __restrict__ img, const float* __restrict__ w,
                     const float* __restrict__ bias, float* __restrict__ f) {
  __shared__ float patch[108];
  __shared__ float psum[27];
  int blk = blockIdx.x;
  int b = blk >> 10;
  int rem = blk & 1023;
  int y = rem >> 5, x = rem & 31;
  int tid = threadIdx.x;
  if (tid < 108) {
    int pos = tid / 27, q = tid % 27;
    int ci = q / 9, ky = (q % 9) / 3, kx = q % 3;
    int Y = 2 * y + (pos >> 1), X = 2 * x + (pos & 1);
    int iy = Y * 16 - 1 + ky, ix = X * 16 - 1 + kx;
    float v = 0.0f;
    if (iy >= 0 && iy < 1024 && ix >= 0 && ix < 1024)
      v = img[(((size_t)b * 3 + ci) * 1024 + iy) * 1024 + ix];
    patch[tid] = v;
  }
  __syncthreads();
  if (tid < 27) psum[tid] = patch[tid] + patch[27 + tid] + patch[54 + tid] + patch[81 + tid];
  __syncthreads();
  int c = tid;
  const float* wr = w + c * 27;
  float acc = 0.0f;
#pragma unroll
  for (int q = 0; q < 27; ++q) acc += wr[q] * psum[q];
  f[(((size_t)b * 256 + c) * 32 + y) * 32 + x] = acc * 0.25f + bias[c];
}

// ------------- RPN 3x3 conv + relu, only at (y=0, x=0..3) ------------------
// Block per (b, x); thread = output channel co (256). ky=0 row is padding.
__global__ void __launch_bounds__(256)
rpn_kernel(const float* __restrict__ f, const float* __restrict__ w,
           const float* __restrict__ bias, float* __restrict__ t) {
  __shared__ float fp[256 * 6];
  int b = blockIdx.x >> 2, x = blockIdx.x & 3;
  int tid = threadIdx.x;
#pragma unroll
  for (int ry = 0; ry < 2; ++ry)
#pragma unroll
    for (int rx = 0; rx < 3; ++rx) {
      int col = x - 1 + rx;
      float v = (col >= 0 && col < 32)
                    ? f[(((size_t)b * 256 + tid) * 32 + ry) * 32 + col] : 0.0f;
      fp[tid * 6 + ry * 3 + rx] = v;
    }
  __syncthreads();
  float acc = bias[tid];
  const float* wr = w + (size_t)tid * 2304;
  for (int ci = 0; ci < 256; ++ci) {
    const float* wc = wr + ci * 9;
    const float* fc = fp + ci * 6;
#pragma unroll
    for (int ky = 1; ky < 3; ++ky)
#pragma unroll
      for (int kx = 0; kx < 3; ++kx)
        acc += wc[ky * 3 + kx] * fc[(ky - 1) * 3 + kx];
  }
  t[((size_t)b * 4 + x) * 256 + tid] = fmaxf(acc, 0.0f);
}

// -------------------- delta 1x1 conv + anchor decode -----------------------
__global__ void decode_kernel(const float* __restrict__ t, const float* __restrict__ dw,
                              const float* __restrict__ db, float* __restrict__ boxes) {
  int p = blockIdx.x * blockDim.x + threadIdx.x;
  if (p >= 800) return;
  int b = p / 100, r = p % 100;
  int x = r / 27, a = r % 27;
  const float* tr = t + ((size_t)b * 4 + x) * 256;
  float d[5];
#pragma unroll
  for (int j = 0; j < 5; ++j) {
    const float* wr = dw + (size_t)(a * 5 + j) * 256;
    float acc = db[a * 5 + j];
    for (int ci = 0; ci < 256; ++ci) acc += tr[ci] * wr[ci];
    d[j] = acc;
  }
  int si = a / 9, ri = (a % 9) / 3, ai = a % 3;
  float scale = (float)(32 << si);
  float sq = (ri == 0) ? 0.70710678118654752f : (ri == 1 ? 1.0f : 1.41421356237309515f);
  float aw = scale * sq, ah = scale / sq;
  boxes[p * 5 + 0] = d[0] * aw + 32.0f * (float)x;
  boxes[p * 5 + 1] = d[1] * ah;                 // acy = 0 (y==0 row)
  boxes[p * 5 + 2] = expf(d[2]) * aw;
  boxes[p * 5 + 3] = expf(d[3]) * ah;
  boxes[p * 5 + 4] = 45.0f * (float)ai + d[4];
}

// ----------------------- rotated ROI align (7x7) ---------------------------
// Block per proposal; thread = channel. Writes pooled activations as bf16
// (GEMM A-operand). Features are 8 MB -> L2-resident gathers.
__global__ void __launch_bounds__(256)
roi_align_kernel(const float* __restrict__ f, const float* __restrict__ boxes,
                 unsigned short* __restrict__ pooled) {
  int p = blockIdx.x;
  int c = threadIdx.x;
  int bi = p / 100;
  float cx = boxes[p * 5 + 0], cy = boxes[p * 5 + 1];
  float w = boxes[p * 5 + 2], h = boxes[p * 5 + 3], ang = boxes[p * 5 + 4];
  float ar = -ang * 0.017453292519943295f;
  float cs = cosf(ar), sn = sinf(ar);
  float t00 =  w * (1.0f / 32.0f) * cs;
  float t01 = -h * (1.0f / 32.0f) * sn;
  float t02 =  cx * (1.0f / 16.0f) - 1.0f;
  float t10 =  w * (1.0f / 32.0f) * sn;
  float t11 =  h * (1.0f / 32.0f) * cs;
  float t12 =  cy * (1.0f / 16.0f) - 1.0f;
  const float* fc = f + ((size_t)bi * 256 + c) * 1024;
  unsigned short* outp = pooled + (size_t)p * 12544 + c * 49;
  for (int s = 0; s < 49; ++s) {
    int i7 = s / 7, j7 = s % 7;
    float lx = (2.0f * j7 + 1.0f) * (1.0f / 7.0f) - 1.0f;
    float ly = (2.0f * i7 + 1.0f) * (1.0f / 7.0f) - 1.0f;
    float gx = t00 * lx + t01 * ly + t02;
    float gy = t10 * lx + t11 * ly + t12;
    float fx = ((gx + 1.0f) * 32.0f - 1.0f) * 0.5f;
    float fy = ((gy + 1.0f) * 32.0f - 1.0f) * 0.5f;
    float x0 = floorf(fx), y0 = floorf(fy);
    float wx = fx - x0, wy = fy - y0;
    int x0i = (int)x0, y0i = (int)y0;
    float v = 0.0f;
#pragma unroll
    for (int dy = 0; dy < 2; ++dy)
#pragma unroll
      for (int dx = 0; dx < 2; ++dx) {
        int xi = x0i + dx, yi = y0i + dy;
        bool valid = (xi >= 0) && (xi < 32) && (yi >= 0) && (yi < 32);
        int xc = min(max(xi, 0), 31), yc = min(max(yi, 0), 31);
        float wt = (dx ? wx : 1.0f - wx) * (dy ? wy : 1.0f - wy);
        v += fc[yc * 32 + xc] * (valid ? wt : 0.0f);
      }
    outp[s] = f32_to_bf16_bits(v);
  }
}

// --------------- bf16 WMMA GEMM, 2x4 register-blocked, bias+relu -----------
// C[M,N] = A[M,K] * B[N,K]^T. One wave computes a 32x64 output block:
// 2 M-fragments x 4 N-fragments -> 8 f32 accumulators. Per 32-wide k-step,
// 6 fragment loads feed 8 independent v_wmma_f32_16x16x32_bf16 (no RAW
// chains, 24 B/lane per WMMA). Fragment layout per ISA 7.12.2: lane l holds
// row/col (l&15); K chunks {k0+8g..+7, k0+16+8g..+7}, g = l>>4.
template <bool OUT_BF16>
__global__ void __launch_bounds__(256)
gemm_bf16_wmma(const __bf16* __restrict__ A, const __bf16* __restrict__ Bw,
               const float* __restrict__ bias, void* __restrict__ out,
               int K, int ntBlk, int totalBlk, int doRelu) {
  int wave = threadIdx.x >> 5;
  int lane = threadIdx.x & 31;
  int blk = blockIdx.x * (blockDim.x >> 5) + wave;
  if (blk >= totalBlk) return;             // wave-uniform; EXEC stays all-1s
  int mb = blk / ntBlk;                    // 32-row block index
  int nb = blk % ntBlk;                    // 64-col block index
  int g = lane >> 4;
  int hl = lane & 15;

  const __bf16* ap0 = A + (size_t)(mb * 32 + hl) * K;
  const __bf16* ap1 = A + (size_t)(mb * 32 + 16 + hl) * K;
  const __bf16* bp0 = Bw + (size_t)(nb * 64 + hl) * K;
  const __bf16* bp1 = Bw + (size_t)(nb * 64 + 16 + hl) * K;
  const __bf16* bp2 = Bw + (size_t)(nb * 64 + 32 + hl) * K;
  const __bf16* bp3 = Bw + (size_t)(nb * 64 + 48 + hl) * K;

  v8f acc[8];
#pragma unroll
  for (int i = 0; i < 8; ++i) acc[i] = (v8f){0.f, 0.f, 0.f, 0.f, 0.f, 0.f, 0.f, 0.f};

  for (int k = 0; k < K; k += 32) {
    int off = k + g * 8;
    v16bf av[2], bv[4];
    {
      v8bf lo, hi;
      lo = *(const v8bf*)(ap0 + off); hi = *(const v8bf*)(ap0 + off + 16);
#pragma unroll
      for (int i = 0; i < 8; ++i) { av[0][i] = lo[i]; av[0][i + 8] = hi[i]; }
      lo = *(const v8bf*)(ap1 + off); hi = *(const v8bf*)(ap1 + off + 16);
#pragma unroll
      for (int i = 0; i < 8; ++i) { av[1][i] = lo[i]; av[1][i + 8] = hi[i]; }
      lo = *(const v8bf*)(bp0 + off); hi = *(const v8bf*)(bp0 + off + 16);
#pragma unroll
      for (int i = 0; i < 8; ++i) { bv[0][i] = lo[i]; bv[0][i + 8] = hi[i]; }
      lo = *(const v8bf*)(bp1 + off); hi = *(const v8bf*)(bp1 + off + 16);
#pragma unroll
      for (int i = 0; i < 8; ++i) { bv[1][i] = lo[i]; bv[1][i + 8] = hi[i]; }
      lo = *(const v8bf*)(bp2 + off); hi = *(const v8bf*)(bp2 + off + 16);
#pragma unroll
      for (int i = 0; i < 8; ++i) { bv[2][i] = lo[i]; bv[2][i + 8] = hi[i]; }
      lo = *(const v8bf*)(bp3 + off); hi = *(const v8bf*)(bp3 + off + 16);
#pragma unroll
      for (int i = 0; i < 8; ++i) { bv[3][i] = lo[i]; bv[3][i + 8] = hi[i]; }
    }
#pragma unroll
    for (int i = 0; i < 2; ++i)
#pragma unroll
      for (int j = 0; j < 4; ++j)
        acc[i * 4 + j] = __builtin_amdgcn_wmma_f32_16x16x32_bf16(
            false, av[i], false, bv[j], (short)0, acc[i * 4 + j], false, false);
  }

  int N = ntBlk * 64;
#pragma unroll
  for (int i = 0; i < 2; ++i)
#pragma unroll
    for (int j = 0; j < 4; ++j) {
      int n = nb * 64 + j * 16 + hl;
      float bval = bias[n];
#pragma unroll
      for (int r = 0; r < 8; ++r) {
        int m = mb * 32 + i * 16 + r + g * 8;  // C/D layout: VGPR r, half g
        float v = acc[i * 4 + j][r] + bval;
        if (doRelu) v = fmaxf(v, 0.0f);
        if (OUT_BF16)
          ((unsigned short*)out)[(size_t)m * N + n] = f32_to_bf16_bits(v);
        else
          ((float*)out)[(size_t)m * N + n] = v;
      }
    }
}

// ----------------------------- cls/bbox heads ------------------------------
__global__ void __launch_bounds__(256)
heads_kernel(const float* __restrict__ x, const float* __restrict__ cw,
             const float* __restrict__ cb, const float* __restrict__ bw,
             const float* __restrict__ bb, float* __restrict__ out) {
  int gid = blockIdx.x * blockDim.x + threadIdx.x;
  int p = gid / 10, j = gid % 10;
  if (p >= 800) return;
  const float* xr = x + (size_t)p * 1024;
  const float* wr;
  float acc;
  int oidx;
  if (j < 5) { wr = cw + j * 1024; acc = cb[j]; oidx = p * 5 + j; }
  else       { wr = bw + (j - 5) * 1024; acc = bb[j - 5]; oidx = 4000 + p * 5 + (j - 5); }
  for (int k = 0; k < 1024; ++k) acc += xr[k] * wr[k];
  out[oidx] = acc;
}

// ------------------------------- launcher ----------------------------------
extern "C" void kernel_launch(void* const* d_in, const int* in_sizes, int n_in,
                              void* d_out, int out_size, void* d_ws, size_t ws_size,
                              hipStream_t stream) {
  const float* images = (const float*)d_in[0];
  const float* bwgt   = (const float*)d_in[1];
  const float* bbias  = (const float*)d_in[2];
  const float* rw     = (const float*)d_in[3];
  const float* rb     = (const float*)d_in[4];
  // d_in[5], d_in[6]: obj head -> dead code in reference
  const float* dw     = (const float*)d_in[7];
  const float* db     = (const float*)d_in[8];
  const float* w1     = (const float*)d_in[9];
  const float* b1     = (const float*)d_in[10];
  const float* w2     = (const float*)d_in[11];
  const float* b2     = (const float*)d_in[12];
  const float* cw     = (const float*)d_in[13];
  const float* cb     = (const float*)d_in[14];
  const float* ow     = (const float*)d_in[15];
  const float* ob     = (const float*)d_in[16];

  // Workspace layout (bytes, 256-aligned), total ~61.2 MB
  char* ws = (char*)d_ws;
  float*          f    = (float*)(ws + 0);                  //  8,388,608
  float*          t    = (float*)(ws + 8388608);            //     32,768
  float*          box  = (float*)(ws + 8421376);            //     16,128
  unsigned short* pool = (unsigned short*)(ws + 8437504);   // 20,070,400
  unsigned short* w1b  = (unsigned short*)(ws + 28507904);  // 25,690,112
  unsigned short* w2b  = (unsigned short*)(ws + 54198016);  //  2,097,152
  unsigned short* x1b  = (unsigned short*)(ws + 56295168);  //  1,638,400
  float*          x2   = (float*)(ws + 57933568);           //  3,276,800

  cvt_bf16_kernel<<<2048, 256, 0, stream>>>(w1, w1b, 1024 * 12544);
  cvt_bf16_kernel<<<512, 256, 0, stream>>>(w2, w2b, 1024 * 1024);
  backbone_pool_kernel<<<8192, 256, 0, stream>>>(images, bwgt, bbias, f);
  rpn_kernel<<<32, 256, 0, stream>>>(f, rw, rb, t);
  decode_kernel<<<7, 128, 0, stream>>>(t, dw, db, box);
  roi_align_kernel<<<800, 256, 0, stream>>>(f, box, pool);
  // fc1: M=800 -> 25 row-blocks of 32; N=1024 -> 16 col-blocks of 64
  //      400 wave-blocks / 8 waves per 256-thread block = 50 blocks
  gemm_bf16_wmma<true><<<50, 256, 0, stream>>>(
      (const __bf16*)pool, (const __bf16*)w1b, b1, (void*)x1b, 12544, 16, 400, 1);
  // fc2: M=800, N=1024, K=1024
  gemm_bf16_wmma<false><<<50, 256, 0, stream>>>(
      (const __bf16*)x1b, (const __bf16*)w2b, b2, (void*)x2, 1024, 16, 400, 1);
  heads_kernel<<<32, 256, 0, stream>>>(x2, cw, cb, ow, ob, (float*)d_out);
}
